// CrossTaskAttention_70557722739027
// MI455X (gfx1250) — matrix operand order repaired and checked by reading, not compiled, not run on hardware
//
#include <hip/hip_runtime.h>
#include <hip/hip_bf16.h>

// ---------------------------------------------------------------------------
// CrossTaskAttention for MI455X (gfx1250, wave32, WMMA).
//
// Math note: seq-len-1 softmax is identically 1, so the Q/K projections are
// dead code; attention collapses to (kv @ wv^T + bv) @ wo^T + bo.
// All GEMMs run on v_wmma_f32_16x16x32_bf16 with fp32 accumulation.
// ---------------------------------------------------------------------------

typedef __attribute__((ext_vector_type(16))) __bf16 v16bf;
typedef __attribute__((ext_vector_type(8)))  float  v8f;

#define E_DIM  1024
#define B_ROWS 16384

#define BM 128
#define BN 128
#define BK 32
#define LDP 40   // padded LDS row stride (elements) -> 80B rows, conflict-free b128

__device__ __forceinline__ unsigned short f2bf(float f) {
  unsigned u = __float_as_uint(f);
  u += 0x7FFFu + ((u >> 16) & 1u);      // round-to-nearest-even
  return (unsigned short)(u >> 16);
}

// ---- fp32 -> bf16 bulk convert (vectorized, n multiple of 4) --------------
__global__ __launch_bounds__(256)
void cvt_f32_bf16(const float* __restrict__ src, unsigned short* __restrict__ dst, int n4) {
  int i = blockIdx.x * 256 + threadIdx.x;
  if (i < n4) {
    float4 f = ((const float4*)src)[i];
    ushort4 h;
    h.x = f2bf(f.x); h.y = f2bf(f.y); h.z = f2bf(f.z); h.w = f2bf(f.w);
    ((ushort4*)dst)[i] = h;
  }
}

// ---- WMMA bf16 GEMM: out = A(MxK) * W(NxK)^T + bias [gelu] [+residual] ----
union FragU { uint4 q[2]; v16bf v; };

__global__ __launch_bounds__(256)
void gemm_bf16_wmma(const unsigned short* __restrict__ A,
                    const unsigned short* __restrict__ W,
                    const float* __restrict__ bias,
                    const float* __restrict__ residual,
                    float* __restrict__ outF,
                    unsigned short* __restrict__ outH,
                    int M, int N, int K, int gelu) {
  __shared__ __align__(16) unsigned short As[BM * LDP];
  __shared__ __align__(16) unsigned short Bs[BN * LDP];

  const int tid   = threadIdx.x;
  const int lane  = tid & 31;
  const int wave  = tid >> 5;
  const int mw    = wave & 3;    // 4 waves along M
  const int nw    = wave >> 2;   // 2 waves along N
  const int lhalf = lane >> 4;   // lane half selects K-chunk / row+8
  const int l15   = lane & 15;

  const int rowBase = blockIdx.y * BM;
  const int colBase = blockIdx.x * BN;

  // cooperative tile load mapping: 256 threads x 16 bf16 = 128x32 tile
  const int ldRow = tid >> 1;
  const int ldSeg = tid & 1;
  const unsigned short* aPtr = A + (size_t)(rowBase + ldRow) * K + ldSeg * 16;
  const unsigned short* wPtr = W + (size_t)(colBase + ldRow) * K + ldSeg * 16;
  unsigned short* asPtr = &As[ldRow * LDP + ldSeg * 16];
  unsigned short* bsPtr = &Bs[ldRow * LDP + ldSeg * 16];

  // ISA 16-bit A-matrix 16x32 layout: lanes 0-15 rows M=l, K={0..7,16..23};
  // lanes 16-31 rows M=l-16, K={8..15,24..31}.
  const int aRow0 = mw * 32 + l15;
  const int aKlo  = lhalf ? 8 : 0;
  const int aKhi  = lhalf ? 24 : 16;
  // B-matrix 32x16: lane(col)=l15; lanes 0-15 hold K={0..7,16..23}, lanes 16-31 K={8..15,24..31}.
  const int bCol0 = nw * 64 + l15;
  const int bK    = lhalf * 8;

  v8f acc[2][4];
#pragma unroll
  for (int i = 0; i < 2; ++i)
#pragma unroll
    for (int j = 0; j < 4; ++j)
      acc[i][j] = (v8f){0.f,0.f,0.f,0.f,0.f,0.f,0.f,0.f};

  for (int k0 = 0; k0 < K; k0 += BK) {
    uint4 a0 = *(const uint4*)(aPtr + k0);
    uint4 a1 = *(const uint4*)(aPtr + k0 + 8);
    uint4 b0 = *(const uint4*)(wPtr + k0);
    uint4 b1 = *(const uint4*)(wPtr + k0 + 8);
    __syncthreads();                       // previous iter's frag reads done
    *(uint4*)asPtr       = a0;
    *(uint4*)(asPtr + 8) = a1;
    *(uint4*)bsPtr       = b0;
    *(uint4*)(bsPtr + 8) = b1;
    __syncthreads();

    FragU af[2];
#pragma unroll
    for (int i = 0; i < 2; ++i) {
      const unsigned short* p = &As[(size_t)(aRow0 + i * 16) * LDP];
      af[i].q[0] = *(const uint4*)(p + aKlo);
      af[i].q[1] = *(const uint4*)(p + aKhi);
    }
#pragma unroll
    for (int j = 0; j < 4; ++j) {
      FragU bf;
      const unsigned short* p = &Bs[(size_t)(bCol0 + j * 16) * LDP + bK];
      bf.q[0] = *(const uint4*)(p);
      bf.q[1] = *(const uint4*)(p + 16);
#pragma unroll
      for (int i = 0; i < 2; ++i)
        acc[i][j] = __builtin_amdgcn_wmma_f32_16x16x32_bf16(
            false, af[i].v, false, bf.v, (short)0, acc[i][j], false, false);
    }
  }

  // Epilogue. C/D layout: VGPR r: lanes 0-15 -> M=r, lanes 16-31 -> M=r+8.
#pragma unroll
  for (int i = 0; i < 2; ++i) {
#pragma unroll
    for (int j = 0; j < 4; ++j) {
#pragma unroll
      for (int r = 0; r < 8; ++r) {
        int rr = rowBase + mw * 32 + i * 16 + r + lhalf * 8;
        int cc = colBase + nw * 64 + j * 16 + l15;
        float v = acc[i][j][r] + bias[cc];
        if (gelu) v = 0.5f * v * (1.0f + erff(v * 0.70710678118654752f));
        size_t idx = (size_t)rr * N + cc;
        if (residual) v += residual[idx];
        if (outF) outF[idx] = v;
        if (outH) outH[idx] = f2bf(v);
      }
    }
  }
}

// ---- fused residual-free LayerNorm: y = (x-m)*rsqrt(var+eps)*g + b --------
__global__ __launch_bounds__(256)
void layernorm_kernel(const float* __restrict__ x, const float* __restrict__ g,
                      const float* __restrict__ b, float* __restrict__ yF,
                      unsigned short* __restrict__ yH) {
  __shared__ float red[256];
  const int row = blockIdx.x;
  const float* xr = x + (size_t)row * E_DIM;
  float v[4];
  float s = 0.f;
#pragma unroll
  for (int i = 0; i < 4; ++i) { v[i] = xr[threadIdx.x + i * 256]; s += v[i]; }
  red[threadIdx.x] = s;
  __syncthreads();
  for (int off = 128; off > 0; off >>= 1) {
    if (threadIdx.x < off) red[threadIdx.x] += red[threadIdx.x + off];
    __syncthreads();
  }
  float mean = red[0] * (1.0f / E_DIM);
  __syncthreads();
  float s2 = 0.f;
#pragma unroll
  for (int i = 0; i < 4; ++i) { float d = v[i] - mean; s2 += d * d; }
  red[threadIdx.x] = s2;
  __syncthreads();
  for (int off = 128; off > 0; off >>= 1) {
    if (threadIdx.x < off) red[threadIdx.x] += red[threadIdx.x + off];
    __syncthreads();
  }
  float inv = rsqrtf(red[0] * (1.0f / E_DIM) + 1e-5f);
#pragma unroll
  for (int i = 0; i < 4; ++i) {
    int c = threadIdx.x + i * 256;
    float y = (v[i] - mean) * inv * g[c] + b[c];
    yF[(size_t)row * E_DIM + c] = y;
    yH[(size_t)row * E_DIM + c] = f2bf(y);
  }
}

// ---------------------------------------------------------------------------
extern "C" void kernel_launch(void* const* d_in, const int* in_sizes, int n_in,
                              void* d_out, int out_size, void* d_ws, size_t ws_size,
                              hipStream_t stream) {
  (void)in_sizes; (void)n_in; (void)out_size; (void)ws_size;

  const float* verb   = (const float*)d_in[0];
  const float* noun   = (const float*)d_in[1];
  const float* v2n_wv = (const float*)d_in[4];
  const float* v2n_wo = (const float*)d_in[5];
  const float* v2n_bv = (const float*)d_in[8];
  const float* v2n_bo = (const float*)d_in[9];
  const float* n2v_wv = (const float*)d_in[12];
  const float* n2v_wo = (const float*)d_in[13];
  const float* n2v_bv = (const float*)d_in[16];
  const float* n2v_bo = (const float*)d_in[17];
  const float* ln_v_g = (const float*)d_in[18];
  const float* ln_v_b = (const float*)d_in[19];
  const float* ln_n_g = (const float*)d_in[20];
  const float* ln_n_b = (const float*)d_in[21];
  const float* fv_w1  = (const float*)d_in[22];
  const float* fv_b1  = (const float*)d_in[23];
  const float* fv_w2  = (const float*)d_in[24];
  const float* fv_b2  = (const float*)d_in[25];
  const float* fn_w1  = (const float*)d_in[26];
  const float* fn_b1  = (const float*)d_in[27];
  const float* fn_w2  = (const float*)d_in[28];
  const float* fn_b2  = (const float*)d_in[29];

  const size_t EE = (size_t)E_DIM * E_DIM;   // 1M
  const size_t BE = (size_t)B_ROWS * E_DIM;  // 16.78M

  // workspace layout (bytes)
  char* p = (char*)d_ws;
  unsigned short* w_wv_v = (unsigned short*)p; p += EE * 2;
  unsigned short* w_wo_v = (unsigned short*)p; p += EE * 2;
  unsigned short* w_wv_n = (unsigned short*)p; p += EE * 2;
  unsigned short* w_wo_n = (unsigned short*)p; p += EE * 2;
  unsigned short* w_w1_v = (unsigned short*)p; p += 2 * EE * 2;
  unsigned short* w_w2_v = (unsigned short*)p; p += 2 * EE * 2;
  unsigned short* w_w1_n = (unsigned short*)p; p += 2 * EE * 2;
  unsigned short* w_w2_n = (unsigned short*)p; p += 2 * EE * 2;
  unsigned short* bf0 = (unsigned short*)p; p += BE * 2;       // phase input act (bf16)
  unsigned short* bf1 = (unsigned short*)p; p += BE * 2;       // v-projection (bf16)
  unsigned short* bf2 = (unsigned short*)p; p += BE * 2;       // post-LN (bf16)
  unsigned short* bfH = (unsigned short*)p; p += 2 * BE * 2;   // ffn hidden (bf16)
  float*          f0  = (float*)p;          p += BE * 4;       // fp32 residual/LN buf

  auto cvt = [&](const float* s, unsigned short* d, size_t n) {
    int n4 = (int)(n / 4);
    cvt_f32_bf16<<<(n4 + 255) / 256, 256, 0, stream>>>(s, d, n4);
  };

  cvt(noun,   bf0,    BE);
  cvt(v2n_wv, w_wv_v, EE);   cvt(v2n_wo, w_wo_v, EE);
  cvt(n2v_wv, w_wv_n, EE);   cvt(n2v_wo, w_wo_n, EE);
  cvt(fv_w1,  w_w1_v, 2*EE); cvt(fv_w2,  w_w2_v, 2*EE);
  cvt(fn_w1,  w_w1_n, 2*EE); cvt(fn_w2,  w_w2_n, 2*EE);

  const dim3 blk(256);
  const dim3 gE (E_DIM / BN,     B_ROWS / BM);   // N=1024 grids
  const dim3 g2E(2 * E_DIM / BN, B_ROWS / BM);   // N=2048 grid

  auto phase = [&](const unsigned short* xin, const float* resid,
                   const unsigned short* wv, const float* bv,
                   const unsigned short* wo, const float* bo,
                   const float* lg, const float* lb,
                   const unsigned short* w1, const float* b1,
                   const unsigned short* w2, const float* b2,
                   float* outF, unsigned short* outH) {
    // v = xin @ wv^T + bv                  (bf16)
    gemm_bf16_wmma<<<gE, blk, 0, stream>>>(xin, wv, bv, nullptr, nullptr, bf1,
                                           B_ROWS, E_DIM, E_DIM, 0);
    // t = resid + v @ wo^T + bo            (fp32)
    gemm_bf16_wmma<<<gE, blk, 0, stream>>>(bf1, wo, bo, resid, f0, nullptr,
                                           B_ROWS, E_DIM, E_DIM, 0);
    // y = LN(t)                            (fp32 in-place + bf16)
    layernorm_kernel<<<B_ROWS, blk, 0, stream>>>(f0, lg, lb, f0, bf2);
    // h = gelu(y @ w1^T + b1)              (bf16)
    gemm_bf16_wmma<<<g2E, blk, 0, stream>>>(bf2, w1, b1, nullptr, nullptr, bfH,
                                            B_ROWS, 2 * E_DIM, E_DIM, 1);
    // out = y + h @ w2^T + b2              (fp32 out, bf16 copy for next phase)
    gemm_bf16_wmma<<<gE, blk, 0, stream>>>(bfH, w2, b2, f0, outF, outH,
                                           B_ROWS, E_DIM, 2 * E_DIM, 0);
  };

  float* outVerb = (float*)d_out;
  float* outNoun = outVerb + BE;

  // verb attends to noun (kv = noun, residual = verb); bf0 reused for verb2 bf16
  phase(bf0, verb, w_wv_v, v2n_bv, w_wo_v, v2n_bo, ln_v_g, ln_v_b,
        w_w1_v, fv_b1, w_w2_v, fv_b2, outVerb, bf0);
  // noun attends to refined verb (kv = verb2 in bf0, residual = noun)
  phase(bf0, noun, w_wv_n, n2v_bv, w_wo_n, n2v_bo, ln_n_g, ln_n_b,
        w_w1_n, fn_b1, w_w2_n, fn_b2, outNoun, nullptr);
}
